// StreamingHead_10952166605126
// MI455X (gfx1250) — matrix operand order repaired and checked by reading, not compiled, not run on hardware
//
#include <hip/hip_runtime.h>
#include <hip/hip_bf16.h>

#define B_     4
#define S_     2048
#define DIN_   4096
#define PROJ_  512
#define MEM_   512
#define START_ 1024
#define T_     1024

typedef __attribute__((ext_vector_type(16))) __bf16 bf16x16;
typedef __attribute__((ext_vector_type(8)))  float  floatx8;

union FragBF {
    bf16x16 v;
    uint4   q[2];
};

static __device__ __forceinline__ floatx8 zero8() {
    floatx8 z;
#pragma unroll
    for (int i = 0; i < 8; ++i) z[i] = 0.0f;
    return z;
}

static __device__ __forceinline__ floatx8 wmma_bf16(bf16x16 a, bf16x16 b, floatx8 c) {
    // D = A(16x32 bf16) * B(32x16 bf16) + C(16x16 f32)
    return __builtin_amdgcn_wmma_f32_16x16x32_bf16(false, a, false, b, (short)0, c, false, false);
}

static __device__ __forceinline__ unsigned short f2bf(float f) {
    union { float f; unsigned int u; } x;
    x.f = f;
    unsigned int r = x.u + 0x7FFFu + ((x.u >> 16) & 1u);  // round-to-nearest-even
    return (unsigned short)(r >> 16);
}

// ---------------------------------------------------------------------------
// elementwise f32 -> bf16 convert
// ---------------------------------------------------------------------------
__global__ void f2bf_kernel(const float* __restrict__ in, unsigned short* __restrict__ out, long n) {
    long i = (long)blockIdx.x * blockDim.x + threadIdx.x;
    long stride = (long)gridDim.x * blockDim.x;
    for (; i < n; i += stride) out[i] = f2bf(in[i]);
}

// f32 (R x C) -> bf16 transposed (C x R)   (for right-multiplied weight matrices)
__global__ void f2bfT_kernel(const float* __restrict__ in, unsigned short* __restrict__ out, int R, int C) {
    long n = (long)R * C;
    long i = (long)blockIdx.x * blockDim.x + threadIdx.x;
    long stride = (long)gridDim.x * blockDim.x;
    for (; i < n; i += stride) {
        int r = (int)(i / C), c = (int)(i % C);
        out[(long)c * R + r] = f2bf(in[i]);
    }
}

// ---------------------------------------------------------------------------
// Generic bf16 WMMA GEMM:  C[M,N] = A[M,K] * B[N,K]^T + bias[N]
// A rows are remapped: grow = (row/rpb)*abstride + arow0 + row%rpb
// Optional f32 C out, optional bf16 C out (optionally per-batch transposed).
// One wave computes a 32(M) x 64(N) tile; 8 waves per block.
// ---------------------------------------------------------------------------
__global__ __launch_bounds__(256) void gemm_bf16_kernel(
    const unsigned short* __restrict__ A, const unsigned short* __restrict__ Bw,
    const float* __restrict__ bias, float* __restrict__ C, unsigned short* __restrict__ Cbf,
    int M, int N, int K, int arow0, int rpb, int abstride, int cbfTrans)
{
    const int lane = threadIdx.x & 31;
    const int wave = threadIdx.x >> 5;
    const int l15  = lane & 15;
    const int hi4  = lane >> 4;
    const int tilesN = N >> 6;
    const int tilesM = M >> 5;
    int gw = blockIdx.x * (blockDim.x >> 5) + wave;
    if (gw >= tilesM * tilesN) return;
    const int mt = gw / tilesN;
    const int nt = gw % tilesN;

    int r0 = mt * 32 + l15;
    int r1 = r0 + 16;
    long r0g = (long)(r0 / rpb) * abstride + arow0 + (r0 % rpb);
    long r1g = (long)(r1 / rpb) * abstride + arow0 + (r1 % rpb);
    const unsigned short* pa0 = A + r0g * K;
    const unsigned short* pa1 = A + r1g * K;
    const unsigned short* pb[4];
#pragma unroll
    for (int j = 0; j < 4; ++j)
        pb[j] = Bw + (long)(nt * 64 + j * 16 + l15) * K;

    floatx8 acc[2][4];
#pragma unroll
    for (int i = 0; i < 2; ++i)
#pragma unroll
        for (int j = 0; j < 4; ++j) acc[i][j] = zero8();

    const int K32 = K >> 5;
    for (int kc = 0; kc < K32; ++kc) {
        const int kb = kc * 32;
        FragBF a0, a1;
        a0.q[0] = *(const uint4*)(pa0 + kb + hi4 * 8);
        a0.q[1] = *(const uint4*)(pa0 + kb + 16 + hi4 * 8);
        a1.q[0] = *(const uint4*)(pa1 + kb + hi4 * 8);
        a1.q[1] = *(const uint4*)(pa1 + kb + 16 + hi4 * 8);
        __builtin_prefetch(pa0 + kb + 64, 0, 3);   // global_prefetch_b8 (next-next A chunk)
#pragma unroll
        for (int j = 0; j < 4; ++j) {
            FragBF b;
            const unsigned short* p = pb[j] + kb + hi4 * 8;
            b.q[0] = *(const uint4*)(p);
            b.q[1] = *(const uint4*)(p + 16);
            acc[0][j] = wmma_bf16(a0.v, b.v, acc[0][j]);
            acc[1][j] = wmma_bf16(a1.v, b.v, acc[1][j]);
        }
    }

#pragma unroll
    for (int i = 0; i < 2; ++i) {
#pragma unroll
        for (int j = 0; j < 4; ++j) {
            const int col = nt * 64 + j * 16 + l15;
            const float bj = bias ? bias[col] : 0.0f;
#pragma unroll
            for (int e = 0; e < 8; ++e) {
                const int row = mt * 32 + i * 16 + e + hi4 * 8;
                const float v = acc[i][j][e] + bj;
                if (C) C[(long)row * N + col] = v;
                if (Cbf) {
                    long idx = cbfTrans ? ((long)(row / rpb) * N + col) * rpb + (row % rpb)
                                        : (long)row * N + col;
                    Cbf[idx] = f2bf(v);
                }
            }
        }
    }
}

// ---------------------------------------------------------------------------
// Flash attention: one block (8 waves) per 16 q-rows; k-tiles of 128.
// qbf/kbf: (B*S, 512) bf16;  vtbf: (B, 512, S) bf16 (transposed so B-frags are
// K-contiguous).  Outputs feat f32 and featbf bf16, both (B*S, 512).
// ---------------------------------------------------------------------------
__global__ __launch_bounds__(256) void attn_kernel(
    const unsigned short* __restrict__ qbf, const unsigned short* __restrict__ kbf,
    const unsigned short* __restrict__ vtbf, float* __restrict__ feat,
    unsigned short* __restrict__ featbf)
{
    __shared__ unsigned short Qt[16 * 512];
    __shared__ float          Sst[16 * 128];
    __shared__ unsigned short Pt[16 * 128];
    __shared__ float mrow[16], lrow[16], crow[16];

    const int b    = blockIdx.y;
    const int q0   = blockIdx.x * 16;
    const int tid  = threadIdx.x;
    const int lane = tid & 31;
    const int wave = tid >> 5;
    const int l15  = lane & 15;
    const int hi4  = lane >> 4;

    {   // cooperative Q tile load (16 consecutive rows are contiguous)
        const uint4* src = (const uint4*)(qbf + (long)(b * S_ + q0) * 512);
        uint4* dst = (uint4*)Qt;
        for (int i = tid; i < 16 * 512 / 8; i += 256) dst[i] = src[i];
    }
    if (tid < 16) { mrow[tid] = -1e30f; lrow[tid] = 0.0f; }

    floatx8 accO[4];
#pragma unroll
    for (int j = 0; j < 4; ++j) accO[j] = zero8();
    __syncthreads();

    const float scale = 1.0f / sqrtf(512.0f + 1e-6f);
    const int ktmax = (q0 + 15) / 128;

    for (int kt = 0; kt <= ktmax; ++kt) {
        const int k0 = kt * 128;
        // ---- S = Q * K^T for this wave's 16 keys (full d reduction) ----
        floatx8 sa = zero8();
        const unsigned short* kbase = kbf + (long)(b * S_ + k0 + wave * 16 + l15) * 512;
#pragma unroll
        for (int kc = 0; kc < 16; ++kc) {
            FragBF a, bb;
            const unsigned short* pq = Qt + l15 * 512 + kc * 32 + hi4 * 8;
            a.q[0] = *(const uint4*)(pq);
            a.q[1] = *(const uint4*)(pq + 16);
            const unsigned short* pk = kbase + kc * 32 + hi4 * 8;
            bb.q[0] = *(const uint4*)(pk);
            bb.q[1] = *(const uint4*)(pk + 16);
            sa = wmma_bf16(a.v, bb.v, sa);
        }
        const int kcol = k0 + wave * 16 + l15;
#pragma unroll
        for (int e = 0; e < 8; ++e) {
            const int row = e + hi4 * 8;
            float sv = sa[e] * scale;
            if (kcol > q0 + row) sv = -1e30f;   // causal mask
            Sst[row * 128 + wave * 16 + l15] = sv;
        }
        __syncthreads();

        // ---- online softmax: wave owns rows 2w, 2w+1 ----
#pragma unroll
        for (int rr = 0; rr < 2; ++rr) {
            const int row = wave * 2 + rr;
            float v0 = Sst[row * 128 + lane];
            float v1 = Sst[row * 128 + lane + 32];
            float v2 = Sst[row * 128 + lane + 64];
            float v3 = Sst[row * 128 + lane + 96];
            float mx = fmaxf(fmaxf(v0, v1), fmaxf(v2, v3));
#pragma unroll
            for (int off = 16; off; off >>= 1) mx = fmaxf(mx, __shfl_xor(mx, off));
            const float mo = mrow[row];
            const float mn = fmaxf(mo, mx);
            float p0 = __expf(v0 - mn), p1 = __expf(v1 - mn);
            float p2 = __expf(v2 - mn), p3 = __expf(v3 - mn);
            Pt[row * 128 + lane]      = f2bf(p0);
            Pt[row * 128 + lane + 32] = f2bf(p1);
            Pt[row * 128 + lane + 64] = f2bf(p2);
            Pt[row * 128 + lane + 96] = f2bf(p3);
            float sum = p0 + p1 + p2 + p3;
#pragma unroll
            for (int off = 16; off; off >>= 1) sum += __shfl_xor(sum, off);
            if (lane == 0) {
                const float c = __expf(mo - mn);
                crow[row] = c;
                lrow[row] = lrow[row] * c + sum;
                mrow[row] = mn;
            }
        }
        __syncthreads();

        // ---- rescale O and accumulate P @ V (wave owns 64 output cols) ----
#pragma unroll
        for (int j = 0; j < 4; ++j)
#pragma unroll
            for (int e = 0; e < 8; ++e) accO[j][e] *= crow[e + hi4 * 8];

#pragma unroll
        for (int kc = 0; kc < 4; ++kc) {
            FragBF a;
            const unsigned short* pp = Pt + l15 * 128 + kc * 32 + hi4 * 8;
            a.q[0] = *(const uint4*)(pp);
            a.q[1] = *(const uint4*)(pp + 16);
#pragma unroll
            for (int j = 0; j < 4; ++j) {
                const int n = wave * 64 + j * 16 + l15;
                const unsigned short* pv = vtbf + (long)(b * 512 + n) * S_ + k0 + kc * 32 + hi4 * 8;
                FragBF bb;
                bb.q[0] = *(const uint4*)(pv);
                bb.q[1] = *(const uint4*)(pv + 16);
                accO[j] = wmma_bf16(a.v, bb.v, accO[j]);
            }
        }
        __syncthreads();
    }

#pragma unroll
    for (int j = 0; j < 4; ++j) {
        const int col = wave * 64 + j * 16 + l15;
#pragma unroll
        for (int e = 0; e < 8; ++e) {
            const int row = e + hi4 * 8;
            const float v = accO[j][e] / lrow[row];
            const long idx = (long)(b * S_ + q0 + row) * 512 + col;
            feat[idx] = v;
            featbf[idx] = f2bf(v);
        }
    }
}

// ---------------------------------------------------------------------------
// Prefix pooling + h0 = tanh(pooled @ p2h_w^T + p2h_b).  One block per batch.
// ---------------------------------------------------------------------------
__global__ __launch_bounds__(256) void pool_kernel(
    const float* __restrict__ feat, const float* __restrict__ scorer_w,
    const float* __restrict__ p2h_w, const float* __restrict__ p2h_b,
    float* __restrict__ h0)
{
    __shared__ float sv[START_];
    __shared__ float pooled[MEM_];
    __shared__ float red[256];
    const int b = blockIdx.x, tid = threadIdx.x;
    const float* fb = feat + (long)b * S_ * 512;

    for (int p = tid; p < START_; p += 256) {
        float acc = 0.0f;
        const float* row = fb + (long)p * 512;
        for (int i = 0; i < 512; ++i) acc += row[i] * scorer_w[i];
        sv[p] = acc;
    }
    __syncthreads();
    float mx = -1e30f;
    for (int p = tid; p < START_; p += 256) mx = fmaxf(mx, sv[p]);
    red[tid] = mx; __syncthreads();
    for (int s = 128; s; s >>= 1) { if (tid < s) red[tid] = fmaxf(red[tid], red[tid + s]); __syncthreads(); }
    const float m = red[0]; __syncthreads();
    float lsum = 0.0f;
    for (int p = tid; p < START_; p += 256) { float e = __expf(sv[p] - m); sv[p] = e; lsum += e; }
    red[tid] = lsum; __syncthreads();
    for (int s = 128; s; s >>= 1) { if (tid < s) red[tid] += red[tid + s]; __syncthreads(); }
    const float inv = 1.0f / red[0]; __syncthreads();

    for (int j = tid; j < MEM_; j += 256) {
        float acc = 0.0f;
        for (int p = 0; p < START_; ++p) acc += sv[p] * fb[(long)p * 512 + j];
        pooled[j] = acc * inv;
    }
    __syncthreads();
    for (int j = tid; j < MEM_; j += 256) {
        float acc = p2h_b[j];
        const float* w = p2h_w + (long)j * MEM_;
        for (int i = 0; i < MEM_; ++i) acc += pooled[i] * w[i];
        h0[b * MEM_ + j] = tanhf(acc);
    }
}

// ---------------------------------------------------------------------------
// GRU scan: one 32-wave workgroup.  h kept in LDS as a 16x512 bf16 A-operand
// (batches 0-3 real, rows 4-15 zero padding).  x-side terms XZ/XR/XH were
// precomputed by GEMM.  Each wave owns 16 state columns; U^T streamed from L2.
// ---------------------------------------------------------------------------
__global__ __launch_bounds__(1024) void scan_kernel(
    const float* __restrict__ XZ, const float* __restrict__ XR, const float* __restrict__ XH,
    const unsigned short* __restrict__ UzT, const unsigned short* __restrict__ UrT,
    const unsigned short* __restrict__ UhT, const float* __restrict__ h0,
    const float* __restrict__ mem_w, const float* __restrict__ mem_b,
    float* __restrict__ out)
{
    __shared__ unsigned short hbf[16 * 512];
    __shared__ unsigned short rhbf[16 * 512];
    __shared__ float hf[4 * 512];
    __shared__ float lslot[2][4];

    const int tid = threadIdx.x, lane = tid & 31, wave = tid >> 5;
    const int l15 = lane & 15, hi4 = lane >> 4;

    for (int i = tid; i < 16 * 512; i += 1024) {
        const int r = i >> 9, c = i & 511;
        hbf[i]  = (r < 4) ? f2bf(h0[r * 512 + c]) : (unsigned short)0;
        rhbf[i] = 0;
    }
    for (int i = tid; i < 4 * 512; i += 1024) hf[i] = h0[i];
    if (tid < 8) lslot[tid >> 2][tid & 3] = 0.0f;
    __syncthreads();

    const float memb = mem_b[0];
    const int j = wave * 16 + l15;                 // this lane's state column
    const float mw = mem_w[j];
    const unsigned short* uz = UzT + (long)j * 512;
    const unsigned short* ur = UrT + (long)j * 512;
    const unsigned short* uh = UhT + (long)j * 512;

    for (int t = 0; t < T_; ++t) {
        // ---- z/r matmuls: h @ Uz, h @ Ur ----
        floatx8 az = zero8(), ar = zero8();
#pragma unroll
        for (int kc = 0; kc < 16; ++kc) {
            FragBF a, bz_, br_;
            const unsigned short* ph = hbf + l15 * 512 + kc * 32 + hi4 * 8;
            a.q[0] = *(const uint4*)(ph);
            a.q[1] = *(const uint4*)(ph + 16);
            const unsigned short* pz = uz + kc * 32 + hi4 * 8;
            bz_.q[0] = *(const uint4*)(pz);
            bz_.q[1] = *(const uint4*)(pz + 16);
            const unsigned short* pr = ur + kc * 32 + hi4 * 8;
            br_.q[0] = *(const uint4*)(pr);
            br_.q[1] = *(const uint4*)(pr + 16);
            az = wmma_bf16(a.v, bz_.v, az);
            ar = wmma_bf16(a.v, br_.v, ar);
        }
        float zs[4], hvs[4];
        if (hi4 == 0) {   // lanes 0-15 hold rows 0-7; batches live in e=0..3
#pragma unroll
            for (int e = 0; e < 4; ++e) {
                const long xi = ((long)(e * T_ + t)) * 512 + j;
                const float z = 1.0f / (1.0f + __expf(-(az[e] + XZ[xi])));
                const float r = 1.0f / (1.0f + __expf(-(ar[e] + XR[xi])));
                const float hv = hf[e * 512 + j];
                zs[e] = z; hvs[e] = hv;
                rhbf[e * 512 + j] = f2bf(r * hv);
            }
        }
        __syncthreads();

        // ---- h_hat matmul: (r*h) @ Uh ----
        floatx8 ah = zero8();
#pragma unroll
        for (int kc = 0; kc < 16; ++kc) {
            FragBF a, bh_;
            const unsigned short* pr = rhbf + l15 * 512 + kc * 32 + hi4 * 8;
            a.q[0] = *(const uint4*)(pr);
            a.q[1] = *(const uint4*)(pr + 16);
            const unsigned short* ph = uh + kc * 32 + hi4 * 8;
            bh_.q[0] = *(const uint4*)(ph);
            bh_.q[1] = *(const uint4*)(ph + 16);
            ah = wmma_bf16(a.v, bh_.v, ah);
        }
        const float dt = (t == 0) ? 0.0f : (1.0f / 1023.0f);
        float pv[4] = {0.0f, 0.0f, 0.0f, 0.0f};
        if (hi4 == 0) {
#pragma unroll
            for (int e = 0; e < 4; ++e) {
                const long xi = ((long)(e * T_ + t)) * 512 + j;
                const float hh = tanhf(ah[e] + XH[xi]);
                const float z = zs[e], hv = hvs[e];
                float hn = (1.0f - z) * hv + z * hh;
                hn = hn + dt * (hn - hv);
                hf[e * 512 + j] = hn;
                hbf[e * 512 + j] = f2bf(hn);
                pv[e] = mw * hn;
            }
        }
#pragma unroll
        for (int e = 0; e < 4; ++e) {
            float v = pv[e];
#pragma unroll
            for (int off = 16; off; off >>= 1) v += __shfl_xor(v, off);
            if (lane == 0) atomicAdd(&lslot[t & 1][e], v);
        }
        __syncthreads();
        if (tid < 4) {
            out[tid * T_ + t] = lslot[t & 1][tid] + memb;
            lslot[(t + 1) & 1][tid] = 0.0f;
        }
    }
}

// ---------------------------------------------------------------------------
extern "C" void kernel_launch(void* const* d_in, const int* in_sizes, int n_in,
                              void* d_out, int out_size, void* d_ws, size_t ws_size,
                              hipStream_t stream) {
    (void)in_sizes; (void)n_in; (void)out_size; (void)ws_size;
    const float* x        = (const float*)d_in[0];
    const float* qw       = (const float*)d_in[2];
    const float* qb       = (const float*)d_in[3];
    const float* kw       = (const float*)d_in[4];
    const float* kb       = (const float*)d_in[5];
    const float* vw       = (const float*)d_in[6];
    const float* vb       = (const float*)d_in[7];
    const float* Wz       = (const float*)d_in[8];
    const float* Uz       = (const float*)d_in[9];
    const float* bz       = (const float*)d_in[10];
    const float* Wr       = (const float*)d_in[11];
    const float* Ur       = (const float*)d_in[12];
    const float* br       = (const float*)d_in[13];
    const float* Wh       = (const float*)d_in[14];
    const float* Uh       = (const float*)d_in[15];
    const float* bh       = (const float*)d_in[16];
    const float* p2h_w    = (const float*)d_in[17];
    const float* p2h_b    = (const float*)d_in[18];
    const float* scorer_w = (const float*)d_in[19];
    const float* mem_w    = (const float*)d_in[20];
    const float* mem_b    = (const float*)d_in[21];
    float* out = (float*)d_out;

    char* ws = (char*)d_ws;
    size_t off = 0;
    auto alloc = [&](size_t bytes) -> void* {
        void* p = ws + off;
        off += (bytes + 255) & ~(size_t)255;
        return p;
    };
    unsigned short* xbf   = (unsigned short*)alloc((size_t)B_ * S_ * DIN_ * 2);
    unsigned short* qwbf  = (unsigned short*)alloc((size_t)PROJ_ * DIN_ * 2);
    unsigned short* kwbf  = (unsigned short*)alloc((size_t)PROJ_ * DIN_ * 2);
    unsigned short* vwbf  = (unsigned short*)alloc((size_t)PROJ_ * DIN_ * 2);
    unsigned short* WzT   = (unsigned short*)alloc((size_t)MEM_ * MEM_ * 2);
    unsigned short* WrT   = (unsigned short*)alloc((size_t)MEM_ * MEM_ * 2);
    unsigned short* WhT   = (unsigned short*)alloc((size_t)MEM_ * MEM_ * 2);
    unsigned short* UzT   = (unsigned short*)alloc((size_t)MEM_ * MEM_ * 2);
    unsigned short* UrT   = (unsigned short*)alloc((size_t)MEM_ * MEM_ * 2);
    unsigned short* UhT   = (unsigned short*)alloc((size_t)MEM_ * MEM_ * 2);
    unsigned short* qbf   = (unsigned short*)alloc((size_t)B_ * S_ * PROJ_ * 2);
    unsigned short* kbf   = (unsigned short*)alloc((size_t)B_ * S_ * PROJ_ * 2);
    unsigned short* vtbf  = (unsigned short*)alloc((size_t)B_ * S_ * PROJ_ * 2);
    float*          feat  = (float*)alloc((size_t)B_ * S_ * PROJ_ * 4);
    unsigned short* fbf   = (unsigned short*)alloc((size_t)B_ * S_ * PROJ_ * 2);
    float*          XZ    = (float*)alloc((size_t)B_ * T_ * MEM_ * 4);
    float*          XR    = (float*)alloc((size_t)B_ * T_ * MEM_ * 4);
    float*          XH    = (float*)alloc((size_t)B_ * T_ * MEM_ * 4);
    float*          h0    = (float*)alloc((size_t)B_ * MEM_ * 4);

    // ---- precision conversion ----
    f2bf_kernel<<<4096, 256, 0, stream>>>(x, xbf, (long)B_ * S_ * DIN_);
    f2bf_kernel<<<1024, 256, 0, stream>>>(qw, qwbf, (long)PROJ_ * DIN_);
    f2bf_kernel<<<1024, 256, 0, stream>>>(kw, kwbf, (long)PROJ_ * DIN_);
    f2bf_kernel<<<1024, 256, 0, stream>>>(vw, vwbf, (long)PROJ_ * DIN_);
    f2bfT_kernel<<<512, 256, 0, stream>>>(Wz, WzT, MEM_, MEM_);
    f2bfT_kernel<<<512, 256, 0, stream>>>(Wr, WrT, MEM_, MEM_);
    f2bfT_kernel<<<512, 256, 0, stream>>>(Wh, WhT, MEM_, MEM_);
    f2bfT_kernel<<<512, 256, 0, stream>>>(Uz, UzT, MEM_, MEM_);
    f2bfT_kernel<<<512, 256, 0, stream>>>(Ur, UrT, MEM_, MEM_);
    f2bfT_kernel<<<512, 256, 0, stream>>>(Uh, UhT, MEM_, MEM_);

    // ---- QKV projections (WMMA). V's bf16 copy stored transposed per batch. ----
    const int qkvBlocks = ((B_ * S_ / 32) * (PROJ_ / 64)) / 8;     // 256
    gemm_bf16_kernel<<<qkvBlocks, 256, 0, stream>>>(xbf, qwbf, qb, nullptr, qbf,
        B_ * S_, PROJ_, DIN_, 0, S_, S_, 0);
    gemm_bf16_kernel<<<qkvBlocks, 256, 0, stream>>>(xbf, kwbf, kb, nullptr, kbf,
        B_ * S_, PROJ_, DIN_, 0, S_, S_, 0);
    gemm_bf16_kernel<<<qkvBlocks, 256, 0, stream>>>(xbf, vwbf, vb, nullptr, vtbf,
        B_ * S_, PROJ_, DIN_, 0, S_, S_, 1);

    // ---- flash attention ----
    attn_kernel<<<dim3(S_ / 16, B_), 256, 0, stream>>>(qbf, kbf, vtbf, feat, fbf);

    // ---- prefix pooling -> h0 ----
    pool_kernel<<<B_, 256, 0, stream>>>(feat, scorer_w, p2h_w, p2h_b, h0);

    // ---- precompute x-side GRU terms on assistant slice (WMMA) ----
    const int xBlocks = ((B_ * T_ / 32) * (MEM_ / 64)) / 8;        // 128
    gemm_bf16_kernel<<<xBlocks, 256, 0, stream>>>(fbf, WzT, bz, XZ, nullptr,
        B_ * T_, MEM_, PROJ_, START_, T_, S_, 0);
    gemm_bf16_kernel<<<xBlocks, 256, 0, stream>>>(fbf, WrT, br, XR, nullptr,
        B_ * T_, MEM_, PROJ_, START_, T_, S_, 0);
    gemm_bf16_kernel<<<xBlocks, 256, 0, stream>>>(fbf, WhT, bh, XH, nullptr,
        B_ * T_, MEM_, PROJ_, START_, T_, S_, 0);

    // ---- sequential GRU scan (single 32-wave workgroup) ----
    scan_kernel<<<1, 1024, 0, stream>>>(XZ, XR, XH, UzT, UrT, UhT, h0, mem_w, mem_b, out);
}
